// ThirdOrderSelfAttention_72868415144104
// MI455X (gfx1250) — compile-verified, matrix operand order checked by reading, not run on hardware
//
#include <hip/hip_runtime.h>
#include <hip/hip_bf16.h>
#include <stdint.h>

typedef __attribute__((ext_vector_type(16))) _Float16 v16h;
typedef __attribute__((ext_vector_type(8)))  _Float16 v8h;
typedef __attribute__((ext_vector_type(8)))  float    v8f;

#define TB 2
#define THD 8
#define TT 192
#define TD 64
#define TC 512
#define TM (TB*TT)          // 384 rows for the projection GEMMs
#define NELEM (TM*TC)       // 196608 = B*H*T*D as well

__device__ __forceinline__ v8f wmma16(v16h a, v16h b, v8f c) {
  return __builtin_amdgcn_wmma_f32_16x16x32_f16(false, a, false, b, (short)0, c,
                                                false, false);
}

__device__ __forceinline__ v16h cat8(v8h lo, v8h hi) {
  v16h r;
#pragma unroll
  for (int i = 0; i < 8; ++i) { r[i] = lo[i]; r[i + 8] = hi[i]; }
  return r;
}

// Load 8 f32 at row+o and 8 f32 at row+o+16, convert to f16 operand vector.
// Matches 16-bit WMMA A/B lane layout: low lanes hold K {0..7,16..23}+base,
// high lanes hold K {8..15,24..31}+base (o already includes half_id*8).
__device__ __forceinline__ v16h load_g16(const float* __restrict__ row, int o) {
  const float4* p0 = (const float4*)(row + o);
  const float4* p1 = (const float4*)(row + o + 16);
  float4 a = p0[0], b = p0[1], c = p1[0], d = p1[1];
  float t[16] = {a.x, a.y, a.z, a.w, b.x, b.y, b.z, b.w,
                 c.x, c.y, c.z, c.w, d.x, d.y, d.z, d.w};
  v16h r;
#pragma unroll
  for (int i = 0; i < 16; ++i) r[i] = (_Float16)t[i];
  return r;
}

// 16x16 output tile of [M,K=512] @ [K=512,N] with B given as rows of W.
// Dual accumulators break the WMMA D->C dependency chain (avoids v_nop pads).
__device__ __forceinline__ v8f gemm_tile_512(const float* __restrict__ arow,
                                             const float* __restrict__ brow,
                                             int half_id) {
  v8f c0 = {}, c1 = {};
#pragma unroll 2
  for (int ks = 0; ks < 16; ks += 2) {
    int oa = ks * 32 + half_id * 8;
    int ob = oa + 32;
    v16h a0 = load_g16(arow, oa);
    v16h b0 = load_g16(brow, oa);
    v16h a1 = load_g16(arow, ob);
    v16h b1 = load_g16(brow, ob);
    c0 = wmma16(a0, b0, c0);
    c1 = wmma16(a1, b1, c1);
  }
  return c0 + c1;
}

// ---------------- Kernel 1: five input projections -------------------------
__global__ __launch_bounds__(32) void proj5_kernel(
    const float* __restrict__ x,
    const float* __restrict__ Wq,  const float* __restrict__ bq,
    const float* __restrict__ Wk0, const float* __restrict__ bk0,
    const float* __restrict__ Wk1, const float* __restrict__ bk1,
    const float* __restrict__ Wv0, const float* __restrict__ bv0,
    const float* __restrict__ Wv1, const float* __restrict__ bv1,
    float* __restrict__ qf, float* __restrict__ k0f, float* __restrict__ v0f,
    _Float16* __restrict__ k1h, _Float16* __restrict__ v1h) {
  int mt = blockIdx.x, nt = blockIdx.y, p = blockIdx.z;
  const float* W; const float* bias;
  switch (p) {
    case 0: W = Wq;  bias = bq;  break;
    case 1: W = Wk0; bias = bk0; break;
    case 2: W = Wk1; bias = bk1; break;
    case 3: W = Wv0; bias = bv0; break;
    default: W = Wv1; bias = bv1; break;
  }
  int lane = threadIdx.x, half_id = lane >> 4, l16 = lane & 15;
  const float* arow = x + (mt * 16 + l16) * TC;   // A row (per lane)
  const float* brow = W + (nt * 16 + l16) * TC;   // B col = W row (y = x W^T)
  v8f c = gemm_tile_512(arow, brow, half_id);

  int col = nt * 16 + l16;            // output feature c_out
  int h = col >> 6, d = col & 63;
  float bs = bias[col];
#pragma unroll
  for (int j = 0; j < 8; ++j) {
    int mm = mt * 16 + half_id * 8 + j;           // C row
    int bb = mm / TT, t = mm % TT;
    float val = c[j] + bs;
    int oidx = ((bb * THD + h) * TT + t) * TD + d;
    switch (p) {
      case 0: qf[oidx]  = val * 0.125f; break;    // fold 1/sqrt(d_k)
      case 1: k0f[oidx] = val;          break;
      case 2: k1h[oidx] = (_Float16)val; break;
      case 3: v0f[oidx] = val;          break;
      default: v1h[oidx] = (_Float16)val; break;
    }
  }
}

// ---------------- Kernel 2: fused trilinear attention ----------------------
// One block (4 waves) per (b,h,k). Each wave handles l-tiles {w, w+4, w+8}.
__global__ __launch_bounds__(128) void attn_kernel(
    const float* __restrict__ qf, const float* __restrict__ k0f,
    const float* __restrict__ v0f, const _Float16* __restrict__ k1h,
    const _Float16* __restrict__ v1h, float* __restrict__ attn) {
  __shared__ __align__(16) _Float16 k1s[TT * 72];    // [m][d], row pad 64->72
  __shared__ __align__(16) _Float16 v1t[TD * 200];   // [d][m], row pad 192->200
  __shared__ __align__(16) _Float16 psc[4 * 16 * 40]; // per-wave P transpose
  __shared__ float outred[4 * 64];

  int idx = blockIdx.x;
  int b = idx / (THD * TT);
  int rem = idx % (THD * TT);
  int h = rem / TT, k = rem % TT;
  int bh = b * THD + h;

  int tid = threadIdx.x, w = tid >> 5, lane = tid & 31;
  int half_id = lane >> 4, l16 = lane & 15;

  const _Float16* k1g = k1h + bh * TT * TD;
  const _Float16* v1g = v1h + bh * TT * TD;

  // --- k1 staging via CDNA5 async global->LDS DMA (16B chunks both sides) ---
  // low 32 bits of a generic pointer to __shared__ == LDS byte offset
  uint32_t k1s_base = (uint32_t)(uintptr_t)(&k1s[0]);
  for (int i = tid; i < (TT * TD) / 8; i += 128) {   // 1536 16-byte chunks
    int m = i >> 3, seg = i & 7;
    uint64_t ga = (uint64_t)(uintptr_t)k1g + (uint64_t)(m * 128 + seg * 16);
    uint32_t la = k1s_base + (uint32_t)(m * 144 + seg * 16);
    asm volatile("global_load_async_to_lds_b128 %0, %1, off"
                 :: "v"(la), "v"(ga) : "memory");
  }
  // --- v1 staged transposed (element scatter, manual) ---
  for (int i = tid; i < (TT * TD) / 8; i += 128) {
    int m = i >> 3, seg = i & 7;
    v8h val = *(const v8h*)(v1g + m * 64 + seg * 8);
#pragma unroll
    for (int j = 0; j < 8; ++j) v1t[(seg * 8 + j) * 200 + m] = val[j];
  }
  asm volatile("s_wait_asynccnt 0" ::: "memory");
  __syncthreads();

  // Per-lane q fragments (4 groups of 8 contiguous d), row k
  const float* qrow = qf + (bh * TT + k) * TD;
  float qg[4][8];
#pragma unroll
  for (int g = 0; g < 4; ++g) {
    const float4* p = (const float4*)(qrow + g * 16 + half_id * 8);
    float4 u0 = p[0], u1 = p[1];
    qg[g][0] = u0.x; qg[g][1] = u0.y; qg[g][2] = u0.z; qg[g][3] = u0.w;
    qg[g][4] = u1.x; qg[g][5] = u1.y; qg[g][6] = u1.z; qg[g][7] = u1.w;
  }

  _Float16* myp = &psc[w * 640];  // [16][40]
  float out_acc[4] = {0.f, 0.f, 0.f, 0.f};

  for (int it = 0; it < 3; ++it) {
    int lt = w + it * 4;                    // l-tile index 0..11
    int l = lt * 16 + l16;

    // A operand: G[l,d] = q[k,d]*k0[l,d], f32 product -> f16
    const float* k0row = k0f + (bh * TT + l) * TD;
    __builtin_prefetch(k0row, 0, 1);
    float kg[4][8];
#pragma unroll
    for (int g = 0; g < 4; ++g) {
      const float4* p = (const float4*)(k0row + g * 16 + half_id * 8);
      float4 u0 = p[0], u1 = p[1];
      kg[g][0] = u0.x; kg[g][1] = u0.y; kg[g][2] = u0.z; kg[g][3] = u0.w;
      kg[g][4] = u1.x; kg[g][5] = u1.y; kg[g][6] = u1.z; kg[g][7] = u1.w;
    }
    v16h aG0, aG1;
#pragma unroll
    for (int j = 0; j < 8; ++j) {
      aG0[j]     = (_Float16)(qg[0][j] * kg[0][j]);
      aG0[j + 8] = (_Float16)(qg[1][j] * kg[1][j]);
      aG1[j]     = (_Float16)(qg[2][j] * kg[2][j]);
      aG1[j + 8] = (_Float16)(qg[3][j] * kg[3][j]);
    }

    // Stage 1: S = G @ K1^T  -> 12 tiles of 16x16 f32 in registers
    v8f s[12];
#pragma unroll
    for (int t = 0; t < 12; ++t) {
      const _Float16* bp = &k1s[(t * 16 + l16) * 72];
      v16h b0 = cat8(*(const v8h*)(bp + half_id * 8),
                     *(const v8h*)(bp + 16 + half_id * 8));
      v16h b1 = cat8(*(const v8h*)(bp + 32 + half_id * 8),
                     *(const v8h*)(bp + 48 + half_id * 8));
      v8f c = {};
      c = wmma16(aG0, b0, c);
      c = wmma16(aG1, b1, c);
      s[t] = c;
    }

    // Row softmax over 192 columns (rows spread across 16-lane half-waves)
    float mx[8], sm[8];
#pragma unroll
    for (int j = 0; j < 8; ++j) {
      float m0 = s[0][j];
#pragma unroll
      for (int t = 1; t < 12; ++t) m0 = fmaxf(m0, s[t][j]);
      mx[j] = m0;
    }
#pragma unroll
    for (int dl = 1; dl <= 8; dl <<= 1)
#pragma unroll
      for (int j = 0; j < 8; ++j) mx[j] = fmaxf(mx[j], __shfl_xor(mx[j], dl, 32));
#pragma unroll
    for (int j = 0; j < 8; ++j) sm[j] = 0.f;
#pragma unroll
    for (int t = 0; t < 12; ++t)
#pragma unroll
      for (int j = 0; j < 8; ++j) {
        float e = __expf(s[t][j] - mx[j]);
        s[t][j] = e;
        sm[j] += e;
      }
#pragma unroll
    for (int dl = 1; dl <= 8; dl <<= 1)
#pragma unroll
      for (int j = 0; j < 8; ++j) sm[j] += __shfl_xor(sm[j], dl, 32);
#pragma unroll
    for (int j = 0; j < 8; ++j) sm[j] = 1.f / sm[j];
#pragma unroll
    for (int t = 0; t < 12; ++t)
#pragma unroll
      for (int j = 0; j < 8; ++j) s[t][j] *= sm[j];

    // Stage 2: U = P @ V1  (K=192 in 6 steps, N=64 in 4 tiles)
    v8f c2[4] = {{}, {}, {}, {}};
#pragma unroll
    for (int ks2 = 0; ks2 < 6; ++ks2) {
      // transpose P columns [32*ks2, 32*ks2+32) from C-layout to A-layout
#pragma unroll
      for (int tt = 0; tt < 2; ++tt) {
        int t = 2 * ks2 + tt;
#pragma unroll
        for (int j = 0; j < 8; ++j) {
          int row = half_id * 8 + j;
          myp[row * 40 + tt * 16 + l16] = (_Float16)s[t][j];
        }
      }
      const _Float16* ap = &myp[l16 * 40];
      v16h aP = cat8(*(const v8h*)(ap + half_id * 8),
                     *(const v8h*)(ap + 16 + half_id * 8));
#pragma unroll
      for (int nt = 0; nt < 4; ++nt) {
        const _Float16* vp = &v1t[(nt * 16 + l16) * 200 + ks2 * 32];
        v16h bV = cat8(*(const v8h*)(vp + half_id * 8),
                       *(const v8h*)(vp + 16 + half_id * 8));
        c2[nt] = wmma16(aP, bV, c2[nt]);
      }
    }

    // out[k,d] += sum over the 16 l-rows of v0[l,d] * U[l,d]
    const float* v0base = v0f + (bh * TT + lt * 16) * TD;
#pragma unroll
    for (int nt = 0; nt < 4; ++nt) {
      int d = nt * 16 + l16;
      float acc = 0.f;
#pragma unroll
      for (int j = 0; j < 8; ++j) {
        int row = half_id * 8 + j;
        acc += v0base[row * TD + d] * c2[nt][j];
      }
      out_acc[nt] += acc;
    }
  }

  // combine half-waves, then waves
#pragma unroll
  for (int nt = 0; nt < 4; ++nt) out_acc[nt] += __shfl_xor(out_acc[nt], 16, 32);
  if (lane < 16) {
#pragma unroll
    for (int nt = 0; nt < 4; ++nt) outred[w * 64 + nt * 16 + l16] = out_acc[nt];
  }
  __syncthreads();
  if (tid < 64) {
    float sum = outred[tid] + outred[64 + tid] + outred[128 + tid] + outred[192 + tid];
    attn[(b * TT + k) * TC + h * TD + tid] = sum;
  }
}

// ---------------- Kernel 3: output projection ------------------------------
__global__ __launch_bounds__(32) void oproj_kernel(
    const float* __restrict__ attn, const float* __restrict__ Wo,
    const float* __restrict__ bo, float* __restrict__ out) {
  int mt = blockIdx.x, nt = blockIdx.y;
  int lane = threadIdx.x, half_id = lane >> 4, l16 = lane & 15;
  const float* arow = attn + (mt * 16 + l16) * TC;
  const float* brow = Wo + (nt * 16 + l16) * TC;
  v8f c = gemm_tile_512(arow, brow, half_id);
  int col = nt * 16 + l16;
  float bs = bo[col];
#pragma unroll
  for (int j = 0; j < 8; ++j) {
    int mm = mt * 16 + half_id * 8 + j;
    out[mm * TC + col] = c[j] + bs;
  }
}

extern "C" void kernel_launch(void* const* d_in, const int* in_sizes, int n_in,
                              void* d_out, int out_size, void* d_ws, size_t ws_size,
                              hipStream_t stream) {
  const float* x   = (const float*)d_in[0];
  const float* Wq  = (const float*)d_in[1];
  const float* bq  = (const float*)d_in[2];
  const float* Wk0 = (const float*)d_in[3];
  const float* bk0 = (const float*)d_in[4];
  const float* Wk1 = (const float*)d_in[5];
  const float* bk1 = (const float*)d_in[6];
  const float* Wv0 = (const float*)d_in[7];
  const float* bv0 = (const float*)d_in[8];
  const float* Wv1 = (const float*)d_in[9];
  const float* bv1 = (const float*)d_in[10];
  const float* Wo  = (const float*)d_in[11];
  const float* bo  = (const float*)d_in[12];
  float* out = (float*)d_out;

  float* qf   = (float*)d_ws;
  float* k0f  = qf  + NELEM;
  float* v0f  = k0f + NELEM;
  float* attn = v0f + NELEM;
  _Float16* k1h = (_Float16*)(attn + NELEM);
  _Float16* v1h = k1h + NELEM;

  // 5 projections: 24 m-tiles x 32 n-tiles x 5 weights, one wave each
  proj5_kernel<<<dim3(24, 32, 5), 32, 0, stream>>>(
      x, Wq, bq, Wk0, bk0, Wk1, bk1, Wv0, bv0, Wv1, bv1,
      qf, k0f, v0f, k1h, v1h);

  // fused trilinear attention: one block per (b,h,k)
  attn_kernel<<<dim3(TB * THD * TT), 128, 0, stream>>>(qf, k0f, v0f, k1h, v1h,
                                                       attn);

  // output projection
  oproj_kernel<<<dim3(24, 32), 32, 0, stream>>>(attn, Wo, bo, out);
}
